// Attention_65249143161365
// MI455X (gfx1250) — compile-verified, matrix operand order checked by reading, not run on hardware
//
#include <hip/hip_runtime.h>
#include <hip/hip_bf16.h>

// ---------------- problem constants ----------------
#define B_  8
#define S_  1024
#define DIM_ 768
#define H_  12
#define HD_ 64
#define BS_ (B_ * S_)          // 8192
#define F3_ (3 * HD_)          // 192
static __device__ __constant__ float kScale = 0.03608439182435161f; // 1/sqrt(768)

typedef __bf16 bf16_t;
typedef __attribute__((ext_vector_type(16))) __bf16 v16bf;
typedef __attribute__((ext_vector_type(8)))  __bf16 v8bf;
typedef __attribute__((ext_vector_type(8)))  float  v8f;

// ---------------- WMMA helpers ----------------
__device__ __forceinline__ v8f wmma_bf16(v16bf a, v16bf b, v8f c) {
  // D = A(16x32 bf16) * B(32x16 bf16) + C(16x16 f32)
  return __builtin_amdgcn_wmma_f32_16x16x32_bf16(
      /*neg_a=*/false, a, /*neg_b=*/false, b,
      /*c_mod=*/(short)0, c, /*reuse_a=*/false, /*reuse_b=*/false);
}

// Per-lane fragment load for the 16-bit 16x32 A layout (and B via transposed
// storage): lane supplies row, kbase = kchunk + 8*(lane>>4); elements are
// [kbase..kbase+7] and [kbase+16..kbase+23], each an aligned 16-byte load.
__device__ __forceinline__ v16bf load_frag(const bf16_t* base,
                                           int stride, int row, int kbase) {
  const v8bf lo = *(const v8bf*)(base + (size_t)row * stride + kbase);
  const v8bf hi = *(const v8bf*)(base + (size_t)row * stride + kbase + 16);
  v16bf r;
#pragma unroll
  for (int i = 0; i < 8; ++i) { r[i] = lo[i]; r[i + 8] = hi[i]; }
  return r;
}

// ---------------- CDNA5 async global->LDS copy (ASYNCcnt path) ----------------
// GVS form: global addr = SGPR64 base + VGPR32 byte offset; vdst VGPR carries
// the LDS byte address. Tracked by ASYNCcnt (see CDNA5 ISA).
__device__ __forceinline__ void async_g2l_b128(unsigned lds_addr,
                                               const void* gbase,
                                               unsigned byte_off) {
  asm volatile("global_load_async_to_lds_b128 %0, %1, %2"
               :: "v"(lds_addr), "v"(byte_off),
                  "s"((unsigned long long)(size_t)gbase)
               : "memory");
}
__device__ __forceinline__ void wait_async0() {
  asm volatile("s_wait_asynccnt 0x0" ::: "memory");
}
__device__ __forceinline__ void wait_ds0() {
  asm volatile("s_wait_dscnt 0x0" ::: "memory");
}

// ---------------- conversion / transpose kernels ----------------
__global__ __launch_bounds__(256) void k_cvt_f32_bf16(const float* __restrict__ in,
                                                      bf16_t* __restrict__ out, int n) {
  int i = blockIdx.x * 256 + threadIdx.x;
  if (i < n) out[i] = (bf16_t)in[i];
}

// Wqkv (H, DIM, 3HD) -> WqkvT (H, 3HD, DIM) bf16
__global__ __launch_bounds__(256) void k_tr_wqkv(const float* __restrict__ w,
                                                 bf16_t* __restrict__ wt) {
  int i = blockIdx.x * 256 + threadIdx.x;
  const int N = H_ * DIM_ * F3_;
  if (i >= N) return;
  int f = i % F3_;
  int t = i / F3_;
  int d = t % DIM_;
  int h = t / DIM_;
  wt[((size_t)h * F3_ + f) * DIM_ + d] = (bf16_t)w[i];
}

// Wo (DIM, DIM) -> WoT (DIM, DIM) transposed bf16
__global__ __launch_bounds__(256) void k_tr_wo(const float* __restrict__ w,
                                               bf16_t* __restrict__ wt) {
  int i = blockIdx.x * 256 + threadIdx.x;
  const int N = DIM_ * DIM_;
  if (i >= N) return;
  int n = i % DIM_;
  int k = i / DIM_;
  wt[(size_t)n * DIM_ + k] = (bf16_t)w[i];
}

// ---------------- QKV projection ----------------
// xb: (BS, DIM) bf16   wqkvt: (H, 3HD, DIM) bf16   bqkv: (H, 3HD) f32
// Q,K: (H, BS, HD) bf16 row-major; Vt: (H, B, HD, S) bf16 (transposed for P*V)
__global__ __launch_bounds__(256) void k_qkv(const bf16_t* __restrict__ xb,
                                             const bf16_t* __restrict__ wqkvt,
                                             const float* __restrict__ bqkv,
                                             bf16_t* __restrict__ Q,
                                             bf16_t* __restrict__ K,
                                             bf16_t* __restrict__ Vt) {
  const int wave = threadIdx.x >> 5;
  const int lane = threadIdx.x & 31;
  const int lr = lane & 15, lh = lane >> 4;
  const int tile = blockIdx.x * 8 + wave;       // 512 * 12 * 12 tiles
  const int ft = tile % 12;                     // N tile (f)
  const int t2 = tile / 12;
  const int h  = t2 % H_;
  const int m  = t2 / H_;                       // M tile (0..511)

  const int row = m * 16 + lr;                  // A row for this lane
  const int n   = ft * 16 + lr;                 // B column (= output col)
  const bf16_t* wh = wqkvt + (size_t)h * F3_ * DIM_;

  v8f c = {};
#pragma unroll 4
  for (int kk = 0; kk < DIM_; kk += 32) {
    // WGP-scope speculative prefetch of the streamed A operand two chunks ahead
    __builtin_prefetch(xb + (size_t)row * DIM_ + kk + 64, 0, 3);
    v16bf a = load_frag(xb, DIM_, row, kk + 8 * lh);
    v16bf b = load_frag(wh, DIM_, n, kk + 8 * lh);
    c = wmma_bf16(a, b, c);
  }
  const float bias = bqkv[h * F3_ + n];
#pragma unroll
  for (int j = 0; j < 8; ++j) {
    const int r = m * 16 + j + 8 * lh;          // global row (b*s)
    const float v = c[j] + bias;
    if (n < HD_) {
      Q[((size_t)h * BS_ + r) * HD_ + n] = (bf16_t)v;
    } else if (n < 2 * HD_) {
      K[((size_t)h * BS_ + r) * HD_ + (n - HD_)] = (bf16_t)v;
    } else {
      const int b = r >> 10, s = r & (S_ - 1);
      Vt[(((size_t)h * B_ + b) * HD_ + (n - 2 * HD_)) * S_ + s] = (bf16_t)v;
    }
  }
}

// ---------------- flash attention ----------------
// One wave per 16-row query tile; 4 waves per block share the same (b,h), so
// the 32-key K/V chunks are staged once per block into LDS with CDNA5 async
// global->LDS copies, then all 4 waves read fragments via ds_load.
__global__ __launch_bounds__(128) void k_attn(const bf16_t* __restrict__ Q,
                                              const bf16_t* __restrict__ K,
                                              const bf16_t* __restrict__ Vt,
                                              bf16_t* __restrict__ O) {
  __shared__ __align__(16) bf16_t kbuf[32 * HD_];      // [key][hd]   4KB
  __shared__ __align__(16) bf16_t vbuf[HD_ * 32];      // [hd][key]   4KB
  __shared__ __align__(16) bf16_t pbuf[4][16 * 32];    // per-wave P  4KB
  const int tid  = threadIdx.x;
  const int wave = tid >> 5;
  const int lane = tid & 31;
  const int lr = lane & 15, lh = lane >> 4;
  const int qt = blockIdx.x * 4 + wave;         // query tile 0..63
  const int b = blockIdx.y, h = blockIdx.z;

  const bf16_t* Qh = Q + ((size_t)h * BS_ + (size_t)b * S_) * HD_;
  const bf16_t* Kh = K + ((size_t)h * BS_ + (size_t)b * S_) * HD_;
  const bf16_t* Vh = Vt + (((size_t)h * B_ + b) * HD_) * S_;  // [HD][S]
  bf16_t* pw = pbuf[wave];

  const unsigned klds = (unsigned)(size_t)&kbuf[0];
  const unsigned vlds = (unsigned)(size_t)&vbuf[0];

  const int qrow = qt * 16 + lr;
  const v16bf aq0 = load_frag(Qh, HD_, qrow, 0 + 8 * lh);   // hd 0..31
  const v16bf aq1 = load_frag(Qh, HD_, qrow, 32 + 8 * lh);  // hd 32..63

  v8f o0 = {}, o1 = {}, o2 = {}, o3 = {};
  float mrow[8], lsum[8];
#pragma unroll
  for (int j = 0; j < 8; ++j) { mrow[j] = -1e30f; lsum[j] = 0.0f; }

  for (int kc = 0; kc < S_; kc += 32) {         // 32-key chunks
    // ---- stage K chunk: 32 contiguous rows of 64 bf16 = 4KB linear copy
    async_g2l_b128(klds + tid * 16u, Kh, (unsigned)(kc * HD_ * 2) + tid * 16u);
    async_g2l_b128(klds + tid * 16u + 2048u, Kh,
                   (unsigned)(kc * HD_ * 2) + tid * 16u + 2048u);
    // ---- stage V chunk: 64 rows (hd) x 64 bytes (32 keys); i = row*4 + chunk
#pragma unroll
    for (int rep = 0; rep < 2; ++rep) {
      const int i = tid + rep * 128;            // 0..255
      const int r = i >> 2, ch = i & 3;
      async_g2l_b128(vlds + (unsigned)i * 16u, Vh,
                     (unsigned)((r * S_ + kc) * 2 + ch * 16));
    }
    wait_async0();
    __syncthreads();

    // ---- scores: two 16x16 tiles over keys [kc,kc+16) and [kc+16,kc+32)
    v8f z0 = {}, z1 = {};
    z0 = wmma_bf16(aq0, load_frag(kbuf, HD_,  0 + lr,  0 + 8 * lh), z0);
    z0 = wmma_bf16(aq1, load_frag(kbuf, HD_,  0 + lr, 32 + 8 * lh), z0);
    z1 = wmma_bf16(aq0, load_frag(kbuf, HD_, 16 + lr,  0 + 8 * lh), z1);
    z1 = wmma_bf16(aq1, load_frag(kbuf, HD_, 16 + lr, 32 + 8 * lh), z1);

    // ---- online softmax (rows live in 16-lane halves; masks 1..8 stay inside)
    float p0[8], p1[8], cor[8];
#pragma unroll
    for (int j = 0; j < 8; ++j) {
      const float v0 = z0[j] * kScale;
      const float v1 = z1[j] * kScale;
      float mx = fmaxf(v0, v1);
#pragma unroll
      for (int msk = 1; msk <= 8; msk <<= 1) mx = fmaxf(mx, __shfl_xor(mx, msk, 32));
      const float mnew = fmaxf(mrow[j], mx);
      cor[j] = __expf(mrow[j] - mnew);
      const float e0 = __expf(v0 - mnew);
      const float e1 = __expf(v1 - mnew);
      float sm = e0 + e1;
#pragma unroll
      for (int msk = 1; msk <= 8; msk <<= 1) sm += __shfl_xor(sm, msk, 32);
      lsum[j] = lsum[j] * cor[j] + sm;
      mrow[j] = mnew;
      p0[j] = e0; p1[j] = e1;
    }
#pragma unroll
    for (int j = 0; j < 8; ++j) {
      o0[j] *= cor[j]; o1[j] *= cor[j]; o2[j] *= cor[j]; o3[j] *= cor[j];
    }

    // ---- P (16x32 f32, C layout) -> LDS -> A fragment (bf16)
    const int rb = 8 * lh;
#pragma unroll
    for (int j = 0; j < 8; ++j) {
      pw[(j + rb) * 32 + lr]      = (bf16_t)p0[j];
      pw[(j + rb) * 32 + lr + 16] = (bf16_t)p1[j];
    }
    wait_ds0();
    const v16bf ap = load_frag(pw, 32, lr, 8 * lh);

    // ---- P * V : 4 output dim tiles of 16, B from vbuf (contiguous over keys)
    o0 = wmma_bf16(ap, load_frag(vbuf, 32,  0 + lr, 8 * lh), o0);
    o1 = wmma_bf16(ap, load_frag(vbuf, 32, 16 + lr, 8 * lh), o1);
    o2 = wmma_bf16(ap, load_frag(vbuf, 32, 32 + lr, 8 * lh), o2);
    o3 = wmma_bf16(ap, load_frag(vbuf, 32, 48 + lr, 8 * lh), o3);

    __syncthreads();   // all waves done reading kbuf/vbuf before next chunk
  }

  // ---- finalize: O[b, s, h*64 + d] bf16
#pragma unroll
  for (int j = 0; j < 8; ++j) {
    const float inv = 1.0f / lsum[j];
    const int r = qt * 16 + j + 8 * lh;
    bf16_t* orow = O + ((size_t)b * S_ + r) * DIM_ + h * HD_;
    orow[ 0 + lr] = (bf16_t)(o0[j] * inv);
    orow[16 + lr] = (bf16_t)(o1[j] * inv);
    orow[32 + lr] = (bf16_t)(o2[j] * inv);
    orow[48 + lr] = (bf16_t)(o3[j] * inv);
  }
}

// ---------------- output projection ----------------
__global__ __launch_bounds__(256) void k_oproj(const bf16_t* __restrict__ Ob,
                                               const bf16_t* __restrict__ wot,
                                               const float* __restrict__ bo,
                                               float* __restrict__ out) {
  const int wave = threadIdx.x >> 5;
  const int lane = threadIdx.x & 31;
  const int lr = lane & 15, lh = lane >> 4;
  const int tile = blockIdx.x * 8 + wave;       // 512 * 48 tiles
  const int nt = tile % 48;
  const int m  = tile / 48;
  const int row = m * 16 + lr;
  const int n   = nt * 16 + lr;

  v8f c = {};
#pragma unroll 4
  for (int kk = 0; kk < DIM_; kk += 32) {
    // WGP-scope speculative prefetch of the streamed A operand two chunks ahead
    __builtin_prefetch(Ob + (size_t)row * DIM_ + kk + 64, 0, 3);
    v16bf a = load_frag(Ob, DIM_, row, kk + 8 * lh);
    v16bf b = load_frag(wot, DIM_, n, kk + 8 * lh);
    c = wmma_bf16(a, b, c);
  }
  const float bias = bo[n];
#pragma unroll
  for (int j = 0; j < 8; ++j) {
    out[((size_t)m * 16 + j + 8 * lh) * DIM_ + n] = c[j] + bias;
  }
}

// ---------------- host launcher ----------------
extern "C" void kernel_launch(void* const* d_in, const int* in_sizes, int n_in,
                              void* d_out, int out_size, void* d_ws, size_t ws_size,
                              hipStream_t stream) {
  const float* x    = (const float*)d_in[0];   // (B,S,DIM)
  const float* Wqkv = (const float*)d_in[1];   // (H,DIM,3HD)
  const float* bqkv = (const float*)d_in[2];   // (H,3HD)
  const float* Wo   = (const float*)d_in[3];   // (DIM,DIM)
  const float* bo   = (const float*)d_in[4];   // (DIM,)
  float* out = (float*)d_out;

  char* ws = (char*)d_ws;
  // byte offsets (all 256-aligned)
  bf16_t* Xb    = (bf16_t*)(ws + 0);                    // 8192*768   = 12,582,912 B
  bf16_t* WqkvT = (bf16_t*)(ws + 12582912);             // 12*192*768 =  3,538,944 B
  bf16_t* WoT   = (bf16_t*)(ws + 16121856);             // 768*768    =  1,179,648 B
  bf16_t* Qb    = (bf16_t*)(ws + 17301504);             // 12*8192*64 = 12,582,912 B
  bf16_t* Kb    = (bf16_t*)(ws + 29884416);             // same
  bf16_t* Vt    = (bf16_t*)(ws + 42467328);             // same (transposed V)
  bf16_t* Ob    = (bf16_t*)(ws + 55050240);             // 8192*768   = 12,582,912 B
  (void)in_sizes; (void)n_in; (void)out_size; (void)ws_size;

  // 1) fp32 -> bf16 conversions / weight transposes
  {
    const int n = BS_ * DIM_;                           // 6,291,456
    k_cvt_f32_bf16<<<(n + 255) / 256, 256, 0, stream>>>(x, Xb, n);
  }
  k_tr_wqkv<<<(H_ * DIM_ * F3_ + 255) / 256, 256, 0, stream>>>(Wqkv, WqkvT);
  k_tr_wo<<<(DIM_ * DIM_ + 255) / 256, 256, 0, stream>>>(Wo, WoT);

  // 2) QKV projection: 512 M-tiles * 12 F-tiles * 12 heads, 8 waves/block
  k_qkv<<<dim3((512 * 12 * 12) / 8), 256, 0, stream>>>(Xb, WqkvT, bqkv, Qb, Kb, Vt);

  // 3) flash attention: 64 query tiles per (b,h), 4 waves/block
  k_attn<<<dim3(16, B_, H_), 128, 0, stream>>>(Qb, Kb, Vt, Ob);

  // 4) output projection: 512 M-tiles * 48 N-tiles, 8 waves/block
  k_oproj<<<dim3((512 * 48) / 8), 256, 0, stream>>>(Ob, WoT, bo, out);
}
